// GAR_Fusion_Net3_1606317769305
// MI455X (gfx1250) — matrix operand order repaired and checked by reading, not compiled, not run on hardware
//
#include <hip/hip_runtime.h>

// ---------------------------------------------------------------------------
// GAR fusion net for MI455X (gfx1250): bf16 WMMA everywhere, flash attention,
// TDM (tensor_load_to_lds) staging of GEMM B-panels with LDS padding.
// ---------------------------------------------------------------------------

typedef __bf16 bf16_t;
typedef __attribute__((ext_vector_type(16))) __bf16 v16bf;
typedef __attribute__((ext_vector_type(8)))  __bf16 v8bf;
typedef __attribute__((ext_vector_type(8)))  float  v8f;
typedef __attribute__((ext_vector_type(4))) unsigned int u32x4;
typedef __attribute__((ext_vector_type(8))) int i32x8;
typedef __attribute__((ext_vector_type(4))) int i32x4;

#define NROWS 4096
#define DDIM  512
#define BPAD_LD 520   // LDS B-panel row stride in bf16 (512 data + 8 pad = 1040B)

// D = A(16x32) * B(32x16) + C, bf16 in / f32 accum
__device__ __forceinline__ v8f wmma_bf16(v16bf a, v16bf b, v8f c) {
  return __builtin_amdgcn_wmma_f32_16x16x32_bf16(false, a, false, b,
                                                 (short)0, c, false, false);
}

// A-fragment (16x32, row-major source, leading dim ld):
// lane t: row = t&15, group g = t>>4;
// elems 0..7  -> K = k0 + g*8 + (0..7)
// elems 8..15 -> K = k0 + 16 + g*8 + (0..7)
__device__ __forceinline__ v16bf load_a_frag(const bf16_t* __restrict__ A,
                                             int ld, int m0, int k0, int lane) {
  int m = m0 + (lane & 15);
  int g = lane >> 4;
  const bf16_t* p = A + m * ld + k0 + g * 8;
  v8bf lo = *(const v8bf*)p;
  v8bf hi = *(const v8bf*)(p + 16);
  v16bf r;
#pragma unroll
  for (int i = 0; i < 8; ++i) { r[i] = lo[i]; r[i + 8] = hi[i]; }
  return r;
}

// B-fragment (32x16) from column-major B, i.e. Bt row-major [N][K], ld = K:
// lane t: col n = t&15, group g = t>>4; elems 0..7 -> K = k0 + g*8 + e, elems 8..15 -> K = k0 + 16 + g*8 + (e-8)
__device__ __forceinline__ v16bf load_b_frag(const bf16_t* __restrict__ Bt,
                                             int ld, int n0, int k0, int lane) {
  int n = n0 + (lane & 15);
  int g = lane >> 4;
  const bf16_t* p = Bt + n * ld + k0 + g * 8;
  v8bf lo = *(const v8bf*)p;
  v8bf hi = *(const v8bf*)(p + 16);
  v16bf r;
#pragma unroll
  for (int i = 0; i < 8; ++i) { r[i] = lo[i]; r[i + 8] = hi[i]; }
  return r;
}

// ---------------------- TDM: stage B panel into LDS ------------------------
// Loads `rows` rows of `K` bf16 (row stride K) starting at Bt into LDS at
// lds_off, inserting 16B of pad after every 1024B row (LDS stride = BPAD_LD).
// One TDM descriptor, tracked by TENSORcnt.
__device__ __forceinline__ void tdm_load_b_panel(const bf16_t* Bt, int K,
                                                 int rows, unsigned int lds_off) {
  unsigned long long gaddr = (unsigned long long)(const void*)Bt;
  u32x4 g0;
  g0[0] = 1u;                                    // count=1 (valid descriptor)
  g0[1] = lds_off;                               // lds_addr (bytes)
  g0[2] = (unsigned int)(gaddr & 0xFFFFFFFFu);   // global_addr[31:0]
  g0[3] = (unsigned int)((gaddr >> 32) & 0x01FFFFFFu) | (2u << 30); // type=2
  unsigned int td0  = (unsigned int)(K / 4);     // row length in 8B units (128)
  unsigned int td1  = (unsigned int)rows;        // rows
  unsigned int ts0l = (unsigned int)(K / 4);     // dim0 stride (units)
  i32x8 g1;
  g1[0] = (int)((3u << 16)      // data_size = 8B
              | (1u << 20)      // pad_enable
              | (7u << 22)      // pad_interval: every 256 DWORDs (=1024B row)
              | (3u << 25));    // pad_amount: 4 DWORDs (=16B)
  g1[1] = (int)((td0 & 0xFFFFu) << 16);          // tensor_dim0[15:0] @bits63:48
  g1[2] = (int)((td0 >> 16) | ((td1 & 0xFFFFu) << 16)); // dim0 hi | dim1 lo
  g1[3] = (int)((td1 >> 16) | (td0 << 16));      // dim1 hi | tile_dim0
  g1[4] = (int)(td1 & 0xFFFFu);                  // tile_dim1 | tile_dim2=0
  g1[5] = (int)ts0l;                             // tensor_dim0_stride[31:0]
  g1[6] = 0;                                     // stride0 hi | stride1 lo
  g1[7] = 0;                                     // stride1 hi (unused, 2D)
  i32x4 z4 = {0, 0, 0, 0};                       // groups 2/3: 2-D tensor
#if __clang_major__ >= 23
  i32x8 z8 = {0, 0, 0, 0, 0, 0, 0, 0};
  __builtin_amdgcn_tensor_load_to_lds(g0, g1, z4, z4, z8, 0);
#else
  __builtin_amdgcn_tensor_load_to_lds(g0, g1, z4, z4, 0);
#endif
}

// --------------------------- elementwise converts ---------------------------

__global__ void k_cvt_bf16(const float* __restrict__ in, bf16_t* __restrict__ out,
                           int n4) {
  int i = blockIdx.x * blockDim.x + threadIdx.x;
  if (i >= n4) return;
  float4 v = ((const float4*)in)[i];
  int o = i * 4;
  out[o + 0] = (bf16_t)v.x;
  out[o + 1] = (bf16_t)v.y;
  out[o + 2] = (bf16_t)v.z;
  out[o + 3] = (bf16_t)v.w;
}

// Wt[n*K + k] = W[k*N + n]  (fp32 -> bf16, transpose so B-frags are contiguous)
__global__ void k_transpose_cvt(const float* __restrict__ W,
                                bf16_t* __restrict__ Wt, int K, int Nn) {
  int idx = blockIdx.x * blockDim.x + threadIdx.x;
  if (idx >= K * Nn) return;
  int n = idx / K;
  int k = idx - n * K;
  Wt[idx] = (bf16_t)W[k * Nn + n];
}

// ------------------------------- WMMA GEMM ---------------------------------
// C = A(MxK, bf16 rowmajor) * B(KxN; given as Bt = B^T rowmajor [N][K])
// B panel (64 x K) for this block staged in LDS by the TDM; A streamed from
// global. 4 waves: wave w owns rows [m0, m0+16), all 64 block cols.
// mode 0: outB[row*N+col] = c*scale          (bf16)
// mode 1: outB[col*M+row] = c                (bf16, transposed store)
// mode 2: outB = relu(c + bias[col])         (bf16)
// mode 3: outF = c + bias[col] + resid[r,c]  (fp32)
__global__ void k_gemm_wmma(const bf16_t* __restrict__ A,
                            const bf16_t* __restrict__ Bt,
                            bf16_t* __restrict__ outB, float* __restrict__ outF,
                            const float* __restrict__ bias,
                            const float* __restrict__ resid,
                            int M, int Nn, int K, float scale, int mode) {
  __shared__ bf16_t Bpanel[64 * BPAD_LD];

  int lane = threadIdx.x & 31;
  int w    = threadIdx.x >> 5;
  int m0 = blockIdx.y * 64 + w * 16;
  int n0 = blockIdx.x * 64;

  // One wave issues the async tensor DMA for the block's B panel.
  if (threadIdx.x < 32) {
    unsigned int lds_off = (unsigned int)(unsigned long long)(void*)&Bpanel[0];
    tdm_load_b_panel(Bt + (size_t)n0 * K, K, 64, lds_off);
    __builtin_amdgcn_s_wait_tensorcnt(0);
  }
  __syncthreads();

  v8f acc[4];
#pragma unroll
  for (int j = 0; j < 4; ++j)
#pragma unroll
    for (int r = 0; r < 8; ++r) acc[j][r] = 0.f;

  for (int k0 = 0; k0 < K; k0 += 32) {
    v16bf a = load_a_frag(A, K, m0, k0, lane);
#pragma unroll
    for (int j = 0; j < 4; ++j) {
      v16bf b = load_b_frag(Bpanel, BPAD_LD, 16 * j, k0, lane);  // ds_load_b128
      acc[j] = wmma_bf16(a, b, acc[j]);
    }
  }

  int rbase = m0 + ((lane >> 4) << 3);
  int cbase = n0 + (lane & 15);
#pragma unroll
  for (int j = 0; j < 4; ++j) {
    int col = cbase + 16 * j;
    float bv = (mode >= 2) ? bias[col] : 0.f;
#pragma unroll
    for (int r = 0; r < 8; ++r) {
      int row = rbase + r;
      float c = acc[j][r] * scale;
      if (mode == 0) {
        outB[row * Nn + col] = (bf16_t)c;
      } else if (mode == 1) {
        outB[col * M + row] = (bf16_t)c;
      } else if (mode == 2) {
        c = fmaxf(c + bv, 0.f);
        outB[row * Nn + col] = (bf16_t)c;
      } else {
        outF[row * Nn + col] = c + bv + resid[row * Nn + col];
      }
    }
  }
}

// --------------------------- flash attention -------------------------------
// scores = Q K^T + bias (Q pre-scaled by 1/sqrt(D)), softmax rows, X = P V.
// Block = 4 waves, owns 16 Q rows. Wave w computes S for key cols
// [w*32, w*32+32) of each 128-wide chunk (S computed once per block),
// shares exp'd P via LDS, and accumulates output cols [w*128, w*128+128).
__global__ void k_attn_wmma(const bf16_t* __restrict__ Qb,
                            const bf16_t* __restrict__ Kb,
                            const bf16_t* __restrict__ Vt,   // [D][N] = V^T
                            const float* __restrict__ biasM, // [N][N]
                            const float* __restrict__ resid, // [N][D] fp32
                            float* __restrict__ Xout,        // [N][D] fp32
                            int useExp) {
  __shared__ bf16_t P_lds[16 * 128];
  __shared__ float s_max[4][16];
  __shared__ float s_sum[4][16];

  int lane = threadIdx.x & 31;
  int w    = threadIdx.x >> 5;
  int hl   = lane >> 4;           // lane half: rows r (hl=0) / r+8 (hl=1)
  int m0   = blockIdx.x * 16;

  // Q fragments for this row block live in registers (16 x v16bf)
  v16bf qf[16];
#pragma unroll
  for (int kk = 0; kk < 16; ++kk)
    qf[kk] = load_a_frag(Qb, DDIM, m0, kk * 32, lane);

  v8f   acc[8];
  float m_run[8], l_run[8];
#pragma unroll
  for (int nt = 0; nt < 8; ++nt)
#pragma unroll
    for (int r = 0; r < 8; ++r) acc[nt][r] = 0.f;
#pragma unroll
  for (int r = 0; r < 8; ++r) { m_run[r] = -1e30f; l_run[r] = 0.f; }

  for (int j = 0; j < 32; ++j) {
    int colbase = j * 128;
    int nq = colbase + w * 32;

    // ---- S slice: two 16x16 tiles, contraction over D=512 ----
    v8f s0, s1;
#pragma unroll
    for (int r = 0; r < 8; ++r) { s0[r] = 0.f; s1[r] = 0.f; }
#pragma unroll
    for (int kk = 0; kk < 16; ++kk) {
      v16bf b0 = load_b_frag(Kb, DDIM, nq,      kk * 32, lane);
      v16bf b1 = load_b_frag(Kb, DDIM, nq + 16, kk * 32, lane);
      s0 = wmma_bf16(qf[kk], b0, s0);
      s1 = wmma_bf16(qf[kk], b1, s1);
    }
    // ---- bias ----
#pragma unroll
    for (int r = 0; r < 8; ++r) {
      int row = m0 + r + hl * 8;
      int c0  = nq + (lane & 15);
      float ba = biasM[row * NROWS + c0];
      float bb = biasM[row * NROWS + c0 + 16];
      if (useExp) {  // exp(-De^2 / (2*sigma^2)), sigma = 10
        ba = __expf(-0.005f * ba * ba);
        bb = __expf(-0.005f * bb * bb);
      }
      s0[r] += ba;
      s1[r] += bb;
    }
    // ---- partial row max (reduce across 16-lane halves) ----
    float pm[8];
#pragma unroll
    for (int r = 0; r < 8; ++r) pm[r] = fmaxf(s0[r], s1[r]);
#pragma unroll
    for (int msk = 1; msk < 16; msk <<= 1)
#pragma unroll
      for (int r = 0; r < 8; ++r) pm[r] = fmaxf(pm[r], __shfl_xor(pm[r], msk));
    if ((lane & 15) == 0)
#pragma unroll
      for (int r = 0; r < 8; ++r) s_max[w][hl * 8 + r] = pm[r];
    __syncthreads();

    // ---- global new max, rescale running state ----
    float mnew[8], alpha[8];
#pragma unroll
    for (int r = 0; r < 8; ++r) {
      int rl = hl * 8 + r;
      float v = m_run[r];
#pragma unroll
      for (int ww = 0; ww < 4; ++ww) v = fmaxf(v, s_max[ww][rl]);
      mnew[r]  = v;
      alpha[r] = __expf(m_run[r] - v);
      m_run[r] = v;
    }
#pragma unroll
    for (int nt = 0; nt < 8; ++nt)
#pragma unroll
      for (int r = 0; r < 8; ++r) acc[nt][r] *= alpha[r];

    // ---- P = exp(S - m), partial row sum ----
    float ps[8];
#pragma unroll
    for (int r = 0; r < 8; ++r) {
      float p0 = __expf(s0[r] - mnew[r]);
      float p1 = __expf(s1[r] - mnew[r]);
      s0[r] = p0; s1[r] = p1;
      ps[r] = p0 + p1;
      l_run[r] *= alpha[r];
    }
#pragma unroll
    for (int msk = 1; msk < 16; msk <<= 1)
#pragma unroll
      for (int r = 0; r < 8; ++r) ps[r] += __shfl_xor(ps[r], msk);
    if ((lane & 15) == 0)
#pragma unroll
      for (int r = 0; r < 8; ++r) s_sum[w][hl * 8 + r] = ps[r];

    // ---- stage P (bf16) in LDS: rows 0..15 x cols 0..127 ----
    int cl = w * 32 + (lane & 15);
#pragma unroll
    for (int r = 0; r < 8; ++r) {
      int rl = hl * 8 + r;
      P_lds[rl * 128 + cl]      = (bf16_t)s0[r];
      P_lds[rl * 128 + cl + 16] = (bf16_t)s1[r];
    }
    __syncthreads();

#pragma unroll
    for (int r = 0; r < 8; ++r) {
      int rl = hl * 8 + r;
      float t = 0.f;
#pragma unroll
      for (int ww = 0; ww < 4; ++ww) t += s_sum[ww][rl];
      l_run[r] += t;
    }

    // ---- acc += P(16x128) @ V(128 x [w*128 .. w*128+128)) ----
#pragma unroll
    for (int kk = 0; kk < 4; ++kk) {
      v16bf a = load_a_frag(P_lds, 128, 0, kk * 32, lane);  // ds_load_b128
      int kg = colbase + kk * 32;
#pragma unroll
      for (int nt = 0; nt < 8; ++nt) {
        v16bf b = load_b_frag(Vt, NROWS, w * 128 + nt * 16, kg, lane);
        acc[nt] = wmma_bf16(a, b, acc[nt]);
      }
    }
  }

  // ---- finalize: X = acc / l + resid ----
#pragma unroll
  for (int r = 0; r < 8; ++r) l_run[r] = 1.f / l_run[r];
#pragma unroll
  for (int nt = 0; nt < 8; ++nt) {
    int col = w * 128 + nt * 16 + (lane & 15);
#pragma unroll
    for (int r = 0; r < 8; ++r) {
      int row = m0 + hl * 8 + r;
      Xout[row * DDIM + col] = acc[nt][r] * l_run[r] + resid[row * DDIM + col];
    }
  }
}

// ------------------------------ layernorm ----------------------------------
// One block (128 threads, 4 waves) per row of 512.
__global__ void k_layernorm(const float* __restrict__ X,
                            const float* __restrict__ gg,
                            const float* __restrict__ bb,
                            float* __restrict__ outF,
                            bf16_t* __restrict__ outB) {
  __shared__ float red[4];
  int row = blockIdx.x;
  int tid = threadIdx.x;
  float4 v = ((const float4*)(X + row * DDIM))[tid];

  float s = v.x + v.y + v.z + v.w;
#pragma unroll
  for (int m = 1; m < 32; m <<= 1) s += __shfl_xor(s, m);
  if ((tid & 31) == 0) red[tid >> 5] = s;
  __syncthreads();
  float mean = (red[0] + red[1] + red[2] + red[3]) * (1.0f / DDIM);
  __syncthreads();

  float dx = v.x - mean, dy = v.y - mean, dz = v.z - mean, dw = v.w - mean;
  float q = dx * dx + dy * dy + dz * dz + dw * dw;
#pragma unroll
  for (int m = 1; m < 32; m <<= 1) q += __shfl_xor(q, m);
  if ((tid & 31) == 0) red[tid >> 5] = q;
  __syncthreads();
  float var  = (red[0] + red[1] + red[2] + red[3]) * (1.0f / DDIM);
  float rstd = rsqrtf(var + 1e-5f);

  int c = tid * 4;
  float o0 = dx * rstd * gg[c + 0] + bb[c + 0];
  float o1 = dy * rstd * gg[c + 1] + bb[c + 1];
  float o2 = dz * rstd * gg[c + 2] + bb[c + 2];
  float o3 = dw * rstd * gg[c + 3] + bb[c + 3];
  if (outF) {
    float4 o = {o0, o1, o2, o3};
    ((float4*)(outF + row * DDIM))[tid] = o;
  }
  if (outB) {
    outB[row * DDIM + c + 0] = (bf16_t)o0;
    outB[row * DDIM + c + 1] = (bf16_t)o1;
    outB[row * DDIM + c + 2] = (bf16_t)o2;
    outB[row * DDIM + c + 3] = (bf16_t)o3;
  }
}

// ------------------------------- launcher ----------------------------------

extern "C" void kernel_launch(void* const* d_in, const int* in_sizes, int n_in,
                              void* d_out, int out_size, void* d_ws, size_t ws_size,
                              hipStream_t stream) {
  const float* R  = (const float*)d_in[0];
  const float* L  = (const float*)d_in[1];
  const float* Dg = (const float*)d_in[2];
  const float* De = (const float*)d_in[3];
  float* out = (float*)d_out;

  // workspace carve-up (~55 MB total; per-branch buffers reused, stream-ordered)
  char*  ws  = (char*)d_ws;
  size_t off = 0;
  auto alloc = [&](size_t bytes) -> void* {
    void* p = ws + off;
    off += (bytes + 255) & ~(size_t)255;
    return p;
  };
  const size_t ND = (size_t)NROWS * DDIM;
  const size_t DD = (size_t)DDIM * DDIM;
  bf16_t* Rbf  = (bf16_t*)alloc(ND * 2);
  bf16_t* Lbf  = (bf16_t*)alloc(ND * 2);
  bf16_t* WtQ  = (bf16_t*)alloc(DD * 2);
  bf16_t* WtK  = (bf16_t*)alloc(DD * 2);
  bf16_t* WtV  = (bf16_t*)alloc(DD * 2);
  bf16_t* Wt1  = (bf16_t*)alloc(DD * 2);
  bf16_t* Wt2  = (bf16_t*)alloc(DD * 2);
  bf16_t* Qbf  = (bf16_t*)alloc(ND * 2);
  bf16_t* Kbf  = (bf16_t*)alloc(ND * 2);
  bf16_t* Vtb  = (bf16_t*)alloc(ND * 2);
  float*  Xf   = (float*)alloc(ND * 4);
  float*  Xlnf = (float*)alloc(ND * 4);
  bf16_t* Xlnb = (bf16_t*)alloc(ND * 2);
  bf16_t* Gbf  = (bf16_t*)alloc(ND * 2);
  float*  X2f  = (float*)alloc(ND * 4);

  const float scaleQ = 0.04419417382415922f;  // 1/sqrt(512)

  // fp32 -> bf16 token features
  int n4 = (int)(ND / 4);
  k_cvt_bf16<<<(n4 + 255) / 256, 256, 0, stream>>>(R, Rbf, n4);
  k_cvt_bf16<<<(n4 + 255) / 256, 256, 0, stream>>>(L, Lbf, n4);

  dim3 gT((int)((DD + 255) / 256));
  dim3 gG(DDIM / 64, NROWS / 64);   // (8, 64), 128 threads = 4 waves

  for (int br = 0; br < 2; ++br) {
    const bf16_t* Qsrc  = br ? Rbf : Lbf;   // R branch: Q from L; L branch: Q from R
    const bf16_t* KVs   = br ? Lbf : Rbf;
    const float*  biasM = br ? Dg  : De;
    const int     useExp = br ? 0 : 1;
    const float*  residF = br ? L : R;
    int base = br ? 15 : 4;
    const float* WQ   = (const float*)d_in[base + 0];
    const float* WK   = (const float*)d_in[base + 1];
    const float* WV   = (const float*)d_in[base + 2];
    const float* ln1g = (const float*)d_in[base + 3];
    const float* ln1b = (const float*)d_in[base + 4];
    const float* w1   = (const float*)d_in[base + 5];
    const float* b1   = (const float*)d_in[base + 6];
    const float* w2   = (const float*)d_in[base + 7];
    const float* b2   = (const float*)d_in[base + 8];
    const float* ln2g = (const float*)d_in[base + 9];
    const float* ln2b = (const float*)d_in[base + 10];
    float* outP = out + (size_t)br * ND;

    // transpose-convert weights (Wt = W^T in bf16)
    k_transpose_cvt<<<gT, 256, 0, stream>>>(WQ, WtQ, DDIM, DDIM);
    k_transpose_cvt<<<gT, 256, 0, stream>>>(WK, WtK, DDIM, DDIM);
    k_transpose_cvt<<<gT, 256, 0, stream>>>(WV, WtV, DDIM, DDIM);
    k_transpose_cvt<<<gT, 256, 0, stream>>>(w1, Wt1, DDIM, DDIM);
    k_transpose_cvt<<<gT, 256, 0, stream>>>(w2, Wt2, DDIM, DDIM);

    // QKV projections (Q pre-scaled by 1/sqrt(D); V stored transposed)
    k_gemm_wmma<<<gG, 128, 0, stream>>>(Qsrc, WtQ, Qbf, nullptr, nullptr, nullptr,
                                        NROWS, DDIM, DDIM, scaleQ, 0);
    k_gemm_wmma<<<gG, 128, 0, stream>>>(KVs, WtK, Kbf, nullptr, nullptr, nullptr,
                                        NROWS, DDIM, DDIM, 1.0f, 0);
    k_gemm_wmma<<<gG, 128, 0, stream>>>(KVs, WtV, Vtb, nullptr, nullptr, nullptr,
                                        NROWS, DDIM, DDIM, 1.0f, 1);

    // flash attention + residual add
    k_attn_wmma<<<NROWS / 16, 128, 0, stream>>>(Qbf, Kbf, Vtb, biasM, residF,
                                                Xf, useExp);

    // LN1 -> fp32 residual + bf16 GEMM input
    k_layernorm<<<NROWS, 128, 0, stream>>>(Xf, ln1g, ln1b, Xlnf, Xlnb);

    // FFN: relu(X @ w1 + b1) @ w2 + b2 + X
    k_gemm_wmma<<<gG, 128, 0, stream>>>(Xlnb, Wt1, Gbf, nullptr, b1, nullptr,
                                        NROWS, DDIM, DDIM, 1.0f, 2);
    k_gemm_wmma<<<gG, 128, 0, stream>>>(Gbf, Wt2, nullptr, X2f, b2, Xlnf,
                                        NROWS, DDIM, DDIM, 1.0f, 3);

    // LN2 -> final output
    k_layernorm<<<NROWS, 128, 0, stream>>>(X2f, ln2g, ln2b, outP, nullptr);
  }
}
